// Differential_76501957476889
// MI455X (gfx1250) — matrix operand order repaired and checked by reading, not compiled, not run on hardware
//
#include <hip/hip_runtime.h>

typedef __attribute__((ext_vector_type(16))) _Float16 v16h;
typedef __attribute__((ext_vector_type(8)))  float    v8f;

#define H_ 16
#define S_ 2048
#define D_ 64
#define KT 32           // keys per tile
#define NWAVE 8
#define QB (16 * NWAVE) // 128 query rows per block
#define LAMBDA_INIT 0.8f
#define NEG_INF (-1.0e30f)

static __device__ __forceinline__ v8f wmma_f16(v16h a, v16h b, v8f c) {
  // emits v_wmma_f32_16x16x32_f16
  return __builtin_amdgcn_wmma_f32_16x16x32_f16(false, a, false, b, (short)0, c,
                                                false, false);
}

// DPP16 ROW_XMASK (0x160|mask): xor-shuffle within each 16-lane row, pure VALU.
// Matches the WMMA C-fragment reduction domain (lanes 0-15 / 16-31 are separate rows).
template <int MASK>
static __device__ __forceinline__ float xmask_max(float x) {
  int s = __builtin_amdgcn_update_dpp(__float_as_int(x), __float_as_int(x),
                                      0x160 | MASK, 0xf, 0xf, true);
  return fmaxf(x, __int_as_float(s));
}
static __device__ __forceinline__ float redmax16(float x) {
  x = xmask_max<1>(x);
  x = xmask_max<2>(x);
  x = xmask_max<4>(x);
  x = xmask_max<8>(x);
  return x;
}

// ---------------- lambda = exp(lq1.lk1) - exp(lq2.lk2) + 0.8 ----------------
__global__ void lambda_kernel(const float* __restrict__ lq1,
                              const float* __restrict__ lq2,
                              const float* __restrict__ lk1,
                              const float* __restrict__ lk2,
                              float* __restrict__ lam_out) {
  int lane = threadIdx.x;
  float d1 = lq1[lane] * lk1[lane] + lq1[lane + 32] * lk1[lane + 32];
  float d2 = lq2[lane] * lk2[lane] + lq2[lane + 32] * lk2[lane + 32];
  for (int m = 16; m >= 1; m >>= 1) {
    d1 += __shfl_xor(d1, m, 32);
    d2 += __shfl_xor(d2, m, 32);
  }
  if (lane == 0) lam_out[0] = __expf(d1) - __expf(d2) + LAMBDA_INIT;
}

// -------- flash attention, one stream (sigma = blockIdx.z: q1k1 or q2k2) ----
__global__ __launch_bounds__(256, 2)
void diffattn_stream_kernel(const float* __restrict__ q,
                            const float* __restrict__ k,
                            const float* __restrict__ v,
                            float* __restrict__ o1,
                            float* __restrict__ o2) {
  __shared__ _Float16 Ksh[KT][D_ + 8];        // [key][dim], f16
  __shared__ _Float16 VshT[D_][KT + 8];       // [dim][key], f16 (transposed)
  __shared__ _Float16 Psh[NWAVE][16][KT + 8]; // per-wave P scratch (C->A relayout)

  const int h     = blockIdx.y;
  const int sigma = blockIdx.z;
  float* outp = (sigma == 0) ? o1 : o2;

  const int wave  = threadIdx.x >> 5;
  const int lane  = threadIdx.x & 31;
  const int l16   = lane & 15;
  const int lhalf = lane >> 4;
  const int q0    = blockIdx.x * QB + wave * 16;

  // ---- load Q tile into A-layout fragments (two K-chunks of 32 dims) ----
  // A fragment, this code's contiguous k order (B uses the same): lane%16 = row M; half j -> K = (lane/16)*16 + j
  v16h aq0, aq1;
  const float* qrow = q + ((size_t)(h * S_ + q0 + l16)) * (2 * D_) + sigma * D_;
#pragma unroll
  for (int j = 0; j < 16; ++j) {
    int kk = (lhalf << 4) + j;
    aq0[j] = (_Float16)qrow[kk];
    aq1[j] = (_Float16)qrow[32 + kk];
  }

  // all-ones B fragment: P(16x32) x ones(32x16) -> per-row sums in C layout
  v16h bones;
#pragma unroll
  for (int j = 0; j < 16; ++j) bones[j] = (_Float16)1.0f;

  v8f o[4] = {};        // 16x64 f32 output accumulator (4 col-chunks of 16)
  float mprev[8], lsum[8];
#pragma unroll
  for (int r = 0; r < 8; ++r) { mprev[r] = NEG_INF; lsum[r] = 0.0f; }

  const float scale = 0.125f; // 1/sqrt(64)

  for (int kt = 0; kt < S_; kt += KT) {
    __syncthreads();
    // cooperative stage: K tile (row major) + V tile (transposed), f32 -> f16
#pragma unroll
    for (int i = 0; i < (KT * D_) / 256; ++i) {
      int idx = threadIdx.x + i * 256;
      int key = idx >> 6, dim = idx & 63;
      Ksh[key][dim]  = (_Float16)k[((size_t)(h * S_ + kt + key)) * (2 * D_) + sigma * D_ + dim];
      VshT[dim][key] = (_Float16)v[((size_t)(h * S_ + kt + key)) * D_ + dim];
    }
    __syncthreads();

    if (kt + KT < S_ && lane == 0) { // hint next tile into cache (global_prefetch_b8)
      __builtin_prefetch(&k[((size_t)(h * S_ + kt + KT + wave * 4)) * (2 * D_) + sigma * D_], 0, 3);
      __builtin_prefetch(&v[((size_t)(h * S_ + kt + KT + wave * 4)) * D_], 0, 3);
    }

    // ---- S = Q * K^T : 16 (q) x 32 (keys), two 16-col fragments ----
    // B fragment, same contiguous k order as A: lane%16 = col n; half j -> row = (lane/16)*16 + j
    v8f s0 = {}, s1 = {};
    {
      v16h b;
#pragma unroll
      for (int j = 0; j < 16; ++j) b[j] = Ksh[l16][(lhalf << 4) + j];
      s0 = wmma_f16(aq0, b, s0);
#pragma unroll
      for (int j = 0; j < 16; ++j) b[j] = Ksh[l16][32 + (lhalf << 4) + j];
      s0 = wmma_f16(aq1, b, s0);
#pragma unroll
      for (int j = 0; j < 16; ++j) b[j] = Ksh[16 + l16][(lhalf << 4) + j];
      s1 = wmma_f16(aq0, b, s1);
#pragma unroll
      for (int j = 0; j < 16; ++j) b[j] = Ksh[16 + l16][32 + (lhalf << 4) + j];
      s1 = wmma_f16(aq1, b, s1);
    }

    // ---- online softmax update (C layout: VGPR r -> row r + 8*(lane/16)) ----
    float alphas[8];
#pragma unroll
    for (int r = 0; r < 8; ++r) {
      float x0 = s0[r] * scale, x1 = s1[r] * scale;
      float mx = redmax16(fmaxf(x0, x1));     // DPP row_xmask reduction, no LDS
      float mn = fmaxf(mprev[r], mx);
      float al = __expf(mprev[r] - mn);
      alphas[r] = al;
      mprev[r] = mn;
      float p0 = __expf(x0 - mn);
      float p1 = __expf(x1 - mn);
      int row = r + (lhalf << 3);
      Psh[wave][row][l16]      = (_Float16)p0;
      Psh[wave][row][16 + l16] = (_Float16)p1;
#pragma unroll
      for (int t = 0; t < 4; ++t) o[t][r] *= al;
    }

    // per-wave LDS slab; same-wave DS ops are in-order -> no block barrier
    v16h ap;
#pragma unroll
    for (int j = 0; j < 16; ++j) {
      int kk = (lhalf << 4) + j;
      ap[j] = Psh[wave][l16][kk];
    }

    // row sums of P via WMMA against all-ones B: rs[r] lands exactly where lsum[r] lives
    {
      v8f rs = {};
      rs = wmma_f16(ap, bones, rs);
#pragma unroll
      for (int r = 0; r < 8; ++r) lsum[r] = lsum[r] * alphas[r] + rs[r];
    }

    // ---- O += P * V : 4 output col-chunks of 16 dims ----
#pragma unroll
    for (int t = 0; t < 4; ++t) {
      v16h bv;
#pragma unroll
      for (int j = 0; j < 16; ++j) bv[j] = VshT[t * 16 + l16][(lhalf << 4) + j];
      o[t] = wmma_f16(ap, bv, o[t]);
    }
  }

  // ---- normalize by softmax denominator, write f32 ----
#pragma unroll
  for (int r = 0; r < 8; ++r) {
    float inv = 1.0f / lsum[r];
    int row = q0 + r + (lhalf << 3);
#pragma unroll
    for (int t = 0; t < 4; ++t)
      outp[((size_t)(h * S_ + row)) * D_ + t * 16 + l16] = o[t][r] * inv;
  }
}

// -------- combine (o1 - lam*o2), per-head layernorm, gamma/beta, *0.2 -------
__global__ __launch_bounds__(256)
void combine_norm_kernel(const float* __restrict__ o1,
                         const float* __restrict__ o2,
                         const float* __restrict__ lam_p,
                         const float* __restrict__ gamma,
                         const float* __restrict__ beta,
                         float* __restrict__ out) {
  const int h = blockIdx.x;
  const float lam = lam_p[0];
  const int N = S_ * D_;
  const size_t base = (size_t)h * N;

  float sum = 0.0f, sq = 0.0f;
  for (int i = threadIdx.x; i < N; i += 256) {
    float c = o1[base + i] - lam * o2[base + i];
    sum += c;
    sq += c * c;
  }
  __shared__ float ssum[NWAVE], ssq[NWAVE];
  for (int m = 16; m >= 1; m >>= 1) {
    sum += __shfl_xor(sum, m, 32);
    sq  += __shfl_xor(sq, m, 32);
  }
  int wv = threadIdx.x >> 5, ln = threadIdx.x & 31;
  if (ln == 0) { ssum[wv] = sum; ssq[wv] = sq; }
  __syncthreads();
  __shared__ float smean, srstd;
  if (threadIdx.x == 0) {
    float ts = 0.0f, tq = 0.0f;
    for (int w = 0; w < NWAVE; ++w) { ts += ssum[w]; tq += ssq[w]; }
    float mean = ts / (float)N;
    float var = tq / (float)N - mean * mean;
    smean = mean;
    srstd = rsqrtf(var + 1e-5f);
  }
  __syncthreads();
  float mean = smean, rstd = srstd;
  for (int i = threadIdx.x; i < N; i += 256) {
    int d = i & 63;
    float c = o1[base + i] - lam * o2[base + i];
    out[base + i] = ((c - mean) * rstd * gamma[h * 64 + d] + beta[h * 64 + d]) *
                    (1.0f - LAMBDA_INIT);
  }
}

extern "C" void kernel_launch(void* const* d_in, const int* in_sizes, int n_in,
                              void* d_out, int out_size, void* d_ws, size_t ws_size,
                              hipStream_t stream) {
  (void)in_sizes; (void)n_in; (void)out_size; (void)ws_size;
  const float* q     = (const float*)d_in[0];
  const float* k     = (const float*)d_in[1];
  const float* v     = (const float*)d_in[2];
  const float* lq1   = (const float*)d_in[3];
  const float* lq2   = (const float*)d_in[4];
  const float* lk1   = (const float*)d_in[5];
  const float* lk2   = (const float*)d_in[6];
  const float* gamma = (const float*)d_in[7];
  const float* beta  = (const float*)d_in[8];

  float* o1  = (float*)d_out;                       // stream-1 attention output
  float* o2  = (float*)d_ws;                        // stream-2 attention output
  float* lam = o2 + (size_t)H_ * S_ * D_;           // single float

  lambda_kernel<<<1, 32, 0, stream>>>(lq1, lq2, lk1, lk2, lam);

  dim3 grid(S_ / QB, H_, 2);
  diffattn_stream_kernel<<<grid, 256, 0, stream>>>(q, k, v, o1, o2);

  combine_norm_kernel<<<H_, 256, 0, stream>>>(o1, o2, lam, gamma, beta, o1);
}
